// MyTripletLoss_73899207295321
// MI455X (gfx1250) — matrix operand order repaired and checked by reading, not compiled, not run on hardware
//
#include <hip/hip_runtime.h>
#include <hip/hip_bf16.h>
#include <math.h>

#define NN 512
#define KK 128
#define NORM_S 3.0f

typedef float     v2f  __attribute__((ext_vector_type(2)));
typedef float     v8f  __attribute__((ext_vector_type(8)));
typedef _Float16  v16h __attribute__((ext_vector_type(16)));

// ---------------------------------------------------------------------------
// Stage 1: f = NORM_S * x / ||x||_row  (also emit f16 copy for fallback path)
// grid = 512 blocks of 128 threads (one block per row)
// ---------------------------------------------------------------------------
__global__ void tl_normalize_kernel(const float* __restrict__ x,
                                    float* __restrict__ f,
                                    _Float16* __restrict__ fh) {
    const int row = blockIdx.x;
    const int tid = threadIdx.x;           // 0..127
    const float v = x[row * KK + tid];

    __shared__ float red[128];
    red[tid] = v * v;
    __syncthreads();
    for (int s = 64; s > 0; s >>= 1) {
        if (tid < s) red[tid] += red[tid + s];
        __syncthreads();
    }
    const float scale = NORM_S / sqrtf(red[0]);
    const float out = v * scale;
    f [row * KK + tid] = out;
    fh[row * KK + tid] = (_Float16)out;
}

// ---------------------------------------------------------------------------
// Stage 2: D = clamp(18 - 2 * f @ f^T, 0)  via WMMA.
// One wave per 16x16 output tile; 32x32 tiles; 8 waves per 256-thread block.
//
// fp32 path: V_WMMA_F32_16X16X4_F32, K stepped by 4 (32 issues / tile).
//   A 16x4 layout (ISA 7.12.2): lane L holds row m = L%16;
//     VGPR0 = A[m][k0], VGPR1 = A[m][k0+1], k0 = (L<16 ? 0 : 2) + 4*kk.
//   B = f^T, so B's (k,n) lane layout reads f rows the same way.
// ---------------------------------------------------------------------------
__global__ void tl_gemm_kernel(const float* __restrict__ f,
                               const _Float16* __restrict__ fh,
                               float* __restrict__ D) {
    const int wave = threadIdx.x >> 5;
    const int lane = threadIdx.x & 31;
    const int tile = blockIdx.x * (blockDim.x >> 5) + wave;   // 0..1023
    const int tm = (tile >> 5) << 4;   // tile row base
    const int tn = (tile & 31) << 4;   // tile col base
    const int m  = lane & 15;
    const int hi = lane >> 4;          // 0 = lanes 0-15, 1 = lanes 16-31

    v8f acc = {};

#if __has_builtin(__builtin_amdgcn_wmma_f32_16x16x4_f32)
    // Exact fp32 Gram matrix via V_WMMA_F32_16X16X4_F32
    const float* arow = f + (tm + m) * KK + (hi << 1);  // k offset 0 or 2
    const float* brow = f + (tn + m) * KK + (hi << 1);
    #pragma unroll 4
    for (int k = 0; k < KK; k += 4) {
        v2f a, b;
        a.x = arow[k]; a.y = arow[k + 1];
        b.x = brow[k]; b.y = brow[k + 1];
        acc = __builtin_amdgcn_wmma_f32_16x16x4_f32(
            /*neg_a=*/false, a, /*neg_b=*/false, b,
            /*c_mod=*/(short)0, acc, /*reuse_a=*/false, /*reuse_b=*/false);
    }
#else
    // Fallback: f16 inputs, V_WMMA_F32_16X16X32_F16 (codegen-confirmed)
    const _Float16* arow = fh + (tm + m) * KK + (hi << 4);   // k offset 0 or 16
    const _Float16* brow = fh + (tn + m) * KK + (hi << 4);   // k offset 0 or 16
    for (int k = 0; k < KK; k += 32) {
        v16h a, b;
        #pragma unroll
        for (int t = 0; t < 8; ++t) {
            a[t]     = arow[k + t];
            a[t + 8] = arow[k + 8 + t];
        }
        #pragma unroll
        for (int t = 0; t < 16; ++t) b[t] = brow[k + t];
        acc = __builtin_amdgcn_wmma_f32_16x16x32_f16(
            false, a, false, b, (short)0, acc, false, false);
    }
#endif

    // C/D layout: VGPR r -> M = r (lanes 0-15) / r+8 (lanes 16-31), N = lane%16
    const int col = tn + m;
    #pragma unroll
    for (int r = 0; r < 8; ++r) {
        const int grow = tm + r + (hi << 3);
        const float d = fmaxf(18.0f - 2.0f * acc[r], 0.0f);
        D[grow * NN + col] = d;
    }
}

// ---------------------------------------------------------------------------
// Stage 3: per-anchor-i masked softplus sum + triplet count.
// grid = 512 blocks (one per i) of 256 threads.
// partials[i] = sum_i, partials[512+i] = count_i
// ---------------------------------------------------------------------------
__global__ void tl_triplet_kernel(const float* __restrict__ D,
                                  const int* __restrict__ labels,
                                  float* __restrict__ partials) {
    const int i   = blockIdx.x;
    const int tid = threadIdx.x;          // 0..255

    __shared__ float Drow[NN];
    __shared__ int   lab[NN];
    for (int t = tid; t < NN; t += 256) {
        Drow[t] = D[i * NN + t];
        lab[t]  = labels[t];
    }
    __syncthreads();

    const int li = lab[i];
    float sum = 0.0f;
    float cnt = 0.0f;
    for (int j = 0; j < NN; ++j) {
        if (j == i || lab[j] != li) continue;   // block-uniform branch
        const float dij = Drow[j];
        for (int k = tid; k < NN; k += 256) {
            if (lab[k] == li) continue;
            const float d  = dij - Drow[k];
            const float ad = fabsf(d);
            // stable softplus: log(1+exp(d)) = max(d,0) + log1p(exp(-|d|))
            sum += fmaxf(d, 0.0f) + log1pf(expf(-ad));
            cnt += 1.0f;
        }
    }

    __shared__ float rs[256];
    __shared__ float rc[256];
    rs[tid] = sum;
    rc[tid] = cnt;
    __syncthreads();
    for (int s = 128; s > 0; s >>= 1) {
        if (tid < s) { rs[tid] += rs[tid + s]; rc[tid] += rc[tid + s]; }
        __syncthreads();
    }
    if (tid == 0) {
        partials[i]      = rs[0];
        partials[NN + i] = rc[0];
    }
}

// ---------------------------------------------------------------------------
// Stage 4: deterministic final reduction, out = sum / count
// ---------------------------------------------------------------------------
__global__ void tl_finalize_kernel(const float* __restrict__ partials,
                                   float* __restrict__ out) {
    const int tid = threadIdx.x;          // 0..255
    __shared__ float rs[256];
    __shared__ float rc[256];
    rs[tid] = partials[tid]      + partials[tid + 256];
    rc[tid] = partials[NN + tid] + partials[NN + tid + 256];
    __syncthreads();
    for (int s = 128; s > 0; s >>= 1) {
        if (tid < s) { rs[tid] += rs[tid + s]; rc[tid] += rc[tid + s]; }
        __syncthreads();
    }
    if (tid == 0) out[0] = rs[0] / rc[0];
}

extern "C" void kernel_launch(void* const* d_in, const int* in_sizes, int n_in,
                              void* d_out, int out_size, void* d_ws, size_t ws_size,
                              hipStream_t stream) {
    const float* x      = (const float*)d_in[0];   // [512,128] fp32
    const int*   labels = (const int*)d_in[1];     // [512] int
    float*       out    = (float*)d_out;           // [1] fp32

    char* ws = (char*)d_ws;
    float*    f        = (float*)ws;                                   // 256 KB
    _Float16* fh       = (_Float16*)(ws + NN * KK * sizeof(float));    // 128 KB
    float*    D        = (float*)(ws + NN * KK * (sizeof(float) + sizeof(_Float16)));
    float*    partials = (float*)((char*)D + NN * NN * sizeof(float)); // 4 KB

    tl_normalize_kernel<<<NN, KK, 0, stream>>>(x, f, fh);
    tl_gemm_kernel<<<(NN / 16) * (NN / 16) / 8, 256, 0, stream>>>(f, fh, D);
    tl_triplet_kernel<<<NN, 256, 0, stream>>>(D, labels, partials);
    tl_finalize_kernel<<<1, 256, 0, stream>>>(partials, out);
}
